// DynamicGate_69561290326694
// MI455X (gfx1250) — matrix operand order repaired and checked by reading, not compiled, not run on hardware
//
#include <hip/hip_runtime.h>
#include <math.h>

typedef __attribute__((ext_vector_type(16))) __bf16 v16bf;
typedef __attribute__((ext_vector_type(8)))  float  v8f;

#define HDIM 2048
#define EDIM 64
#define NEGFILL (-1e9f)

// ---------------------------------------------------------------------------
// Kernel 1: transpose sim_matrix [H,E] -> Wt [E,H] in bf16, and compute
// per-expert inverse column norms (1 / max(||w_e||, eps)).
// ---------------------------------------------------------------------------
__global__ __launch_bounds__(256) void dg_prep(const float* __restrict__ W,
                                               __bf16* __restrict__ Wt,
                                               float* __restrict__ invw) {
  const int e = blockIdx.x;   // expert 0..63
  const int t = threadIdx.x;  // 0..255
  float ss = 0.f;
  for (int h = t; h < HDIM; h += 256) {
    float v = W[(size_t)h * EDIM + e];
    ss += v * v;
    Wt[(size_t)e * HDIM + h] = (__bf16)v;
  }
  __shared__ float red[256];
  red[t] = ss;
  __syncthreads();
  for (int off = 128; off > 0; off >>= 1) {
    if (t < off) red[t] += red[t + off];
    __syncthreads();
  }
  if (t == 0) invw[e] = 1.f / fmaxf(sqrtf(red[0]), 1e-12f);
}

// ---------------------------------------------------------------------------
// Kernel 2: fused normalize + GEMM (bf16 WMMA, f32 accum) + gating epilogue.
// One wave handles a 16-token x 64-expert tile. Requires Ntok % 128 == 0.
// ---------------------------------------------------------------------------
__global__ __launch_bounds__(256) void dg_main(
    const float* __restrict__ x, const __bf16* __restrict__ Wt,
    const float* __restrict__ invw, const float* __restrict__ gates,
    float* __restrict__ out_mask, float* __restrict__ out_probs,
    float* __restrict__ out_logits, int Ntok) {
  __shared__ float s_thr[EDIM];
  __shared__ float s_invw[EDIM];
  __shared__ float tile[8][16 * 66];  // per-wave 16x64 tile, padded stride 66

  const int tid = threadIdx.x;
  if (tid < EDIM) {
    s_thr[tid]  = 1.f / (1.f + __expf(-gates[tid]));  // sigmoid(gates)
    s_invw[tid] = invw[tid];
  }
  __syncthreads();

  const int wv   = tid >> 5;
  const int lane = tid & 31;
  const int tile_base = (blockIdx.x * 8 + wv) * 16;

  // A-fragment addressing (contiguous k order, same as B):
  // lanes 0-15: row = lane,   K 0..15 contiguous
  // lanes 16-31: row = lane-16, K 16..31 contiguous
  const int mrow = lane & 15;
  const int kA = (lane & 16) ? 16 : 0;
  // B-fragment addressing (contiguous k order, same as A): lane holds 16 contiguous K
  // of column et*16 + (lane&15), at K offset 0 (lanes 0-15) or 16 (16-31).
  const int kB = (lane & 16) ? 16 : 0;

  const float*  px  = x  + (size_t)(tile_base + mrow) * HDIM + kA;
  const __bf16* pwb = Wt + (size_t)mrow * HDIM + kB;

  v8f acc0 = {0.f,0.f,0.f,0.f,0.f,0.f,0.f,0.f};
  v8f acc1 = acc0, acc2 = acc0, acc3 = acc0;
  float ssq = 0.f;  // partial sum of squares of this lane's half-row of x

  for (int kc = 0; kc < 64; ++kc) {
    const float* p = px + kc * 32;
    float4 f0 = *(const float4*)(p);
    float4 f1 = *(const float4*)(p + 4);
    float4 f2 = *(const float4*)(p + 8);
    float4 f3 = *(const float4*)(p + 12);

    ssq += f0.x*f0.x + f0.y*f0.y + f0.z*f0.z + f0.w*f0.w;
    ssq += f1.x*f1.x + f1.y*f1.y + f1.z*f1.z + f1.w*f1.w;
    ssq += f2.x*f2.x + f2.y*f2.y + f2.z*f2.z + f2.w*f2.w;
    ssq += f3.x*f3.x + f3.y*f3.y + f3.z*f3.z + f3.w*f3.w;

    v16bf a;
    a[0]  = (__bf16)f0.x; a[1]  = (__bf16)f0.y; a[2]  = (__bf16)f0.z; a[3]  = (__bf16)f0.w;
    a[4]  = (__bf16)f1.x; a[5]  = (__bf16)f1.y; a[6]  = (__bf16)f1.z; a[7]  = (__bf16)f1.w;
    a[8]  = (__bf16)f2.x; a[9]  = (__bf16)f2.y; a[10] = (__bf16)f2.z; a[11] = (__bf16)f2.w;
    a[12] = (__bf16)f3.x; a[13] = (__bf16)f3.y; a[14] = (__bf16)f3.z; a[15] = (__bf16)f3.w;

    const __bf16* pw = pwb + kc * 32;
    v16bf b0 = *(const v16bf*)(pw);
    v16bf b1 = *(const v16bf*)(pw + 16 * HDIM);
    v16bf b2 = *(const v16bf*)(pw + 32 * HDIM);
    v16bf b3 = *(const v16bf*)(pw + 48 * HDIM);

    acc0 = __builtin_amdgcn_wmma_f32_16x16x32_bf16(false, a, false, b0, (short)0, acc0, false, false);
    acc1 = __builtin_amdgcn_wmma_f32_16x16x32_bf16(false, a, false, b1, (short)0, acc1, false, false);
    acc2 = __builtin_amdgcn_wmma_f32_16x16x32_bf16(false, a, false, b2, (short)0, acc2, false, false);
    acc3 = __builtin_amdgcn_wmma_f32_16x16x32_bf16(false, a, false, b3, (short)0, acc3, false, false);
  }

  // Complete the row sums of squares: lane l + lane l^16 cover row (l&15).
  float ssfull = ssq + __shfl_xor(ssq, 16, 32);

  // Spill accumulators (raw dot products) to the wave's LDS tile.
  // C/D layout: VGPR r -> M = r + (lane>=16 ? 8 : 0), N = lane & 15.
  float* T = &tile[wv][0];
  const int rbase = (lane & 16) ? 8 : 0;
  const int ecol  = lane & 15;
#pragma unroll
  for (int r = 0; r < 8; ++r) {
    T[(rbase + r) * 66 +  0 + ecol] = acc0[r];
    T[(rbase + r) * 66 + 16 + ecol] = acc1[r];
    T[(rbase + r) * 66 + 32 + ecol] = acc2[r];
    T[(rbase + r) * 66 + 48 + ecol] = acc3[r];
  }
  __syncthreads();

  // ---- Row-wise epilogue: 2 lanes per token, 32 experts each ----
  const int row     = lane >> 1;
  const int halfsel = lane & 1;
  const float ssr  = __shfl(ssfull, row, 32);  // lane 'row' holds row 'row'
  const float srow = 1.f / fmaxf(sqrtf(ssr), 1e-12f);
  float* Tr = T + row * 66 + halfsel * 32;
  const size_t ng    = (size_t)(tile_base + row);
  const int    ebase = halfsel * 32;

  // Pass A: scale logits, write them out, count active, argmax(logits).
  float mymax = -3.402823466e38f;
  int   myidx = 0x7fffffff;
  int   cnt   = 0;
#pragma unroll
  for (int j = 0; j < 32; ++j) {
    const int e = ebase + j;
    float lg = Tr[j] * srow * s_invw[e];
    Tr[j] = lg;  // keep scaled logit in LDS for later passes
    out_logits[ng * EDIM + e] = lg;
    float g = lg - s_thr[e];
    cnt += (g > 0.f) ? 1 : 0;
    if (lg > mymax) { mymax = lg; myidx = e; }  // first max wins on ties
  }
  {
    int   ocnt = __shfl_xor(cnt, 1, 32);
    float omax = __shfl_xor(mymax, 1, 32);
    int   oidx = __shfl_xor(myidx, 1, 32);
    if (omax > mymax || (omax == mymax && oidx < myidx)) { mymax = omax; myidx = oidx; }
    cnt += ocnt;
  }
  const bool forced = (cnt == 0);

  // Pass B: row max of masked gated values.
  float rmax = -3.402823466e38f;
#pragma unroll
  for (int j = 0; j < 32; ++j) {
    const int e = ebase + j;
    float g = Tr[j] - s_thr[e];
    bool  m = (g > 0.f) || (forced && (e == myidx));
    float mv = m ? fmaxf(g, 0.f) : NEGFILL;
    rmax = fmaxf(rmax, mv);
  }
  rmax = fmaxf(rmax, __shfl_xor(rmax, 1, 32));

  // Pass C: softmax denominator.
  float ssum = 0.f;
#pragma unroll
  for (int j = 0; j < 32; ++j) {
    const int e = ebase + j;
    float g = Tr[j] - s_thr[e];
    bool  m = (g > 0.f) || (forced && (e == myidx));
    float mv = m ? fmaxf(g, 0.f) : NEGFILL;
    ssum += __expf(mv - rmax);
  }
  ssum += __shfl_xor(ssum, 1, 32);
  const float invs = 1.f / ssum;

  // Pass D: write probs and mask.
#pragma unroll
  for (int j = 0; j < 32; ++j) {
    const int e = ebase + j;
    float g = Tr[j] - s_thr[e];
    bool  m = (g > 0.f) || (forced && (e == myidx));
    float mv = m ? fmaxf(g, 0.f) : NEGFILL;
    out_probs[ng * EDIM + e] = __expf(mv - rmax) * invs;
    out_mask[ng * EDIM + e]  = m ? 1.f : 0.f;
  }
}

// ---------------------------------------------------------------------------
extern "C" void kernel_launch(void* const* d_in, const int* in_sizes, int n_in,
                              void* d_out, int out_size, void* d_ws, size_t ws_size,
                              hipStream_t stream) {
  const float* x     = (const float*)d_in[0];  // [N, 2048]
  const float* W     = (const float*)d_in[1];  // [2048, 64]
  const float* gates = (const float*)d_in[2];  // [64]
  const int Ntok = in_sizes[0] / HDIM;         // 16384

  float* out        = (float*)d_out;
  float* out_mask   = out;
  float* out_probs  = out + (size_t)Ntok * EDIM;
  float* out_logits = out + 2 * (size_t)Ntok * EDIM;

  __bf16* Wt   = (__bf16*)d_ws;                                   // 256 KB
  float*  invw = (float*)((char*)d_ws + (size_t)EDIM * HDIM * 2); // 256 B

  dg_prep<<<EDIM, 256, 0, stream>>>(W, Wt, invw);
  dg_main<<<Ntok / 128, 256, 0, stream>>>(x, Wt, invw, gates,
                                          out_mask, out_probs, out_logits, Ntok);
}
